// PartialEncoderWeightedSumEDDIMultiWeight_57767310131607
// MI455X (gfx1250) — matrix operand order repaired and checked by reading, not compiled, not run on hardware
//
#include <hip/hip_runtime.h>
#include <stdint.h>

#define CB 64
#define CJ 16384
#define CD 32
#define CH 128
#define CEH 256
#define CL 32
#define CW 4

typedef __attribute__((ext_vector_type(16))) _Float16 v16h;
typedef __attribute__((ext_vector_type(8)))  _Float16 v8h;
typedef __attribute__((ext_vector_type(4)))  _Float16 v4h;
typedef __attribute__((ext_vector_type(8)))  float    v8f;

__device__ __forceinline__ v8f wmma16(v16h a, v16h b, v8f c) {
  // D(16x16 f32) = A(16x32 f16) * B(32x16 f16) + C
  return __builtin_amdgcn_wmma_f32_16x16x32_f16(false, a, false, b, (short)0, c,
                                                false, false);
}

__device__ __forceinline__ float wred_add(float v) {
  v += __shfl_xor(v, 1, 32);  v += __shfl_xor(v, 2, 32);
  v += __shfl_xor(v, 4, 32);  v += __shfl_xor(v, 8, 32);
  v += __shfl_xor(v, 16, 32); return v;
}
__device__ __forceinline__ float wred_max(float v) {
  v = fmaxf(v, __shfl_xor(v, 1, 32));  v = fmaxf(v, __shfl_xor(v, 2, 32));
  v = fmaxf(v, __shfl_xor(v, 4, 32));  v = fmaxf(v, __shfl_xor(v, 8, 32));
  v = fmaxf(v, __shfl_xor(v, 16, 32)); return v;
}
// reduction within each 16-lane half (xor masks < 16 never cross halves)
__device__ __forceinline__ float hred_add(float v) {
  v += __shfl_xor(v, 1, 32); v += __shfl_xor(v, 2, 32);
  v += __shfl_xor(v, 4, 32); v += __shfl_xor(v, 8, 32); return v;
}

// ---------------------------------------------------------------------------
// K1: P[j][h] = feat_emb[j] @ h_w1[1:,:] + h_b1   (J x 128 fp32, L2-resident)
// ---------------------------------------------------------------------------
__global__ __launch_bounds__(256) void k1_proj(const float* __restrict__ feat_emb,
                                               const float* __restrict__ h_w1,
                                               const float* __restrict__ h_b1,
                                               float* __restrict__ P) {
  const int lane = threadIdx.x & 31, wave = threadIdx.x >> 5;
  const int j0  = blockIdx.x * 128 + wave * 16;
  const int nlo = lane & 15;
  const int hih = lane >> 4;

  // A fragment (16x32 f16): lanes 0-15: K 0-15, lanes 16-31: K 16-31
  v16h a;
  const float* ar = feat_emb + (size_t)(j0 + nlo) * CD;
  const int ka = hih * 16;
#pragma unroll
  for (int i = 0; i < 8; ++i) {
    a[i]     = (_Float16)ar[ka + i];
    a[8 + i] = (_Float16)ar[8 + ka + i];
  }
  const int kb = hih * 16;  // B: elements 0..15 are K 0..15 (lanes<16) / 16..31
#pragma unroll
  for (int nc = 0; nc < 8; ++nc) {
    const int n = nc * 16 + nlo;
    v16h bf;
#pragma unroll
    for (int i = 0; i < 16; ++i)
      bf[i] = (_Float16)h_w1[(size_t)(1 + kb + i) * CH + n];
    const float bias = h_b1[n];
    v8f c;
#pragma unroll
    for (int v = 0; v < 8; ++v) c[v] = bias;
    c = wmma16(a, bf, c);
    const int mo = hih * 8;  // C: M = v + (lane>=16 ? 8 : 0), N = lane&15
#pragma unroll
    for (int v = 0; v < 8; ++v)
      P[(size_t)(j0 + mo + v) * CH + n] = c[v];
  }
}

// ---------------------------------------------------------------------------
// K2: per-row fused  h -> h_out -> logits   (dominant kernel)
// ---------------------------------------------------------------------------
__global__ __launch_bounds__(256) void k2_rows(
    const float* __restrict__ x, const int* __restrict__ mask,
    const float* __restrict__ P, const float* __restrict__ h_w1,
    const float* __restrict__ ln1g, const float* __restrict__ ln1b,
    const float* __restrict__ h_w2, const float* __restrict__ h_b2,
    const float* __restrict__ ln2g, const float* __restrict__ ln2b,
    const float* __restrict__ g_w1, const float* __restrict__ g_b1,
    const float* __restrict__ g_w2, const float* __restrict__ g_b2,
    _Float16* __restrict__ houtG, float* __restrict__ logitsG) {
  __shared__ float s_w1r0[CH], s_g1[CH], s_b1[CH];
  __shared__ float s_gw1[CD * 16], s_gb1[16], s_gw2[16 * CW], s_gb2[CW];
  __shared__ _Float16 s_w2h[CH * CD];       // h_w2 in f16, row-major (k*32+n)
  __shared__ _Float16 s_A[8][16 * CH];      // per-wave WMMA A staging
  __shared__ _Float16 s_ho[8][16 * CD];     // per-wave h_out tile

  const int t = threadIdx.x, lane = t & 31, wave = t >> 5;
  const int b = blockIdx.y;
  const int jt = blockIdx.x * 128 + wave * 16;

  if (t < CH) { s_w1r0[t] = h_w1[t]; s_g1[t] = ln1g[t]; s_b1[t] = ln1b[t]; }
  for (int i = t; i < CD * 16; i += 256) s_gw1[i] = g_w1[i];
  if (t < 16) s_gb1[t] = g_b1[t];
  if (t < 64) s_gw2[t] = g_w2[t];
  if (t < 4)  s_gb2[t] = g_b2[t];
  for (int i = t; i < CH * CD; i += 256) s_w2h[i] = (_Float16)h_w2[i];
  __syncthreads();

  __builtin_prefetch(P + (size_t)jt * CH, 0, 1);

  const int nlo = lane & 15;
  const int hih = lane >> 4;
  _Float16* Arow = s_A[wave];

  // ---- stage 1: h = relu(LN(x*w1row0 + P)) for 16 rows, f16 -> LDS -------
  for (int r = 0; r < 16; ++r) {
    const int j = jt + r;
    const float xv = x[(size_t)b * CJ + j];
    float4 p  = *(const float4*)(P + (size_t)j * CH + lane * 4);
    float4 w0 = *(const float4*)(s_w1r0 + lane * 4);
    float v0 = fmaf(xv, w0.x, p.x), v1 = fmaf(xv, w0.y, p.y);
    float v2 = fmaf(xv, w0.z, p.z), v3 = fmaf(xv, w0.w, p.w);
    float s  = wred_add(v0 + v1 + v2 + v3);
    float s2 = wred_add(v0 * v0 + v1 * v1 + v2 * v2 + v3 * v3);
    float mean = s * (1.f / CH);
    float rstd = rsqrtf(s2 * (1.f / CH) - mean * mean + 1e-5f);
    float4 g  = *(const float4*)(s_g1 + lane * 4);
    float4 bb = *(const float4*)(s_b1 + lane * 4);
    v4h hv;
    hv[0] = (_Float16)fmaxf((v0 - mean) * rstd * g.x + bb.x, 0.f);
    hv[1] = (_Float16)fmaxf((v1 - mean) * rstd * g.y + bb.y, 0.f);
    hv[2] = (_Float16)fmaxf((v2 - mean) * rstd * g.z + bb.z, 0.f);
    hv[3] = (_Float16)fmaxf((v3 - mean) * rstd * g.w + bb.w, 0.f);
    *(v4h*)(Arow + r * CH + lane * 4) = hv;
  }
  asm volatile("s_wait_dscnt 0" ::: "memory");

  // ---- stage 2: (16x128) @ h_w2(128x32) via 8 x v_wmma_f32_16x16x32_f16 --
  v8f c0 = {}; v8f c1 = {};
  const int ka = hih * 16;
  const int kb = hih * 16;
#pragma unroll
  for (int kc = 0; kc < 4; ++kc) {
    const _Float16* ap = Arow + nlo * CH + kc * 32 + ka;
    v8h lo = *(const v8h*)ap;
    v8h hh = *(const v8h*)(ap + 8);
    v16h a;
#pragma unroll
    for (int i = 0; i < 8; ++i) { a[i] = lo[i]; a[8 + i] = hh[i]; }
    v16h b0, b1;
#pragma unroll
    for (int i = 0; i < 16; ++i) {
      const int k = kc * 32 + kb + i;
      b0[i] = s_w2h[k * CD + nlo];
      b1[i] = s_w2h[k * CD + 16 + nlo];
    }
    c0 = wmma16(a, b0, c0);
    c1 = wmma16(a, b1, c1);
  }

  // ---- stage 3: + b2, LN(32), relu, store h_out (f16 global + LDS) -------
  const float hb2a = h_b2[nlo],    hb2b = h_b2[16 + nlo];
  const float g2a  = ln2g[nlo],    g2b  = ln2g[16 + nlo];
  const float q2a  = ln2b[nlo],    q2b  = ln2b[16 + nlo];
  const int mo = hih * 8;
#pragma unroll
  for (int v = 0; v < 8; ++v) {
    float a0 = c0[v] + hb2a;
    float a1 = c1[v] + hb2b;
    float s  = hred_add(a0 + a1);
    float s2 = hred_add(a0 * a0 + a1 * a1);
    float mean = s * (1.f / CD);
    float rstd = rsqrtf(s2 * (1.f / CD) - mean * mean + 1e-5f);
    float o0 = fmaxf((a0 - mean) * rstd * g2a + q2a, 0.f);
    float o1 = fmaxf((a1 - mean) * rstd * g2b + q2b, 0.f);
    const int j = jt + mo + v;
    houtG[((size_t)b * CJ + j) * CD + nlo]      = (_Float16)o0;
    houtG[((size_t)b * CJ + j) * CD + 16 + nlo] = (_Float16)o1;
    s_ho[wave][(mo + v) * CD + nlo]      = (_Float16)o0;
    s_ho[wave][(mo + v) * CD + 16 + nlo] = (_Float16)o1;
  }
  asm volatile("s_wait_dscnt 0" ::: "memory");

  // ---- stage 4: g-MLP + clip + mask, lanes 0..15 own one row each --------
  if (lane < 16) {
    const int r = lane, j = jt + r;
    float hr[CD];
#pragma unroll
    for (int d = 0; d < CD; ++d) hr[d] = (float)s_ho[wave][r * CD + d];
    float r0 = s_gb2[0], r1 = s_gb2[1], r2 = s_gb2[2], r3 = s_gb2[3];
#pragma unroll
    for (int i = 0; i < 16; ++i) {
      float tv = s_gb1[i];
#pragma unroll
      for (int d = 0; d < CD; ++d) tv = fmaf(hr[d], s_gw1[d * 16 + i], tv);
      tv = fmaxf(tv, 0.f);
      r0 = fmaf(tv, s_gw2[i * 4 + 0], r0);
      r1 = fmaf(tv, s_gw2[i * 4 + 1], r1);
      r2 = fmaf(tv, s_gw2[i * 4 + 2], r2);
      r3 = fmaf(tv, s_gw2[i * 4 + 3], r3);
    }
    const int mk = mask[(size_t)b * CJ + j];
    float4 lg;
    lg.x = mk ? fminf(fmaxf(r0, -10.f), 10.f) : -1e9f;
    lg.y = mk ? fminf(fmaxf(r1, -10.f), 10.f) : -1e9f;
    lg.z = mk ? fminf(fmaxf(r2, -10.f), 10.f) : -1e9f;
    lg.w = mk ? fminf(fmaxf(r3, -10.f), 10.f) : -1e9f;
    *(float4*)(logitsG + ((size_t)b * CJ + j) * 4) = lg;
  }
}

// ---------------------------------------------------------------------------
// K3: per-b masked softmax + head_sums = E^T(4xJ) @ h_out(Jx32) via WMMA
//     with async global->LDS staging of h_out chunks
// ---------------------------------------------------------------------------
__global__ __launch_bounds__(256) void k3_reduce(const float* __restrict__ logitsG,
                                                 const _Float16* __restrict__ houtG,
                                                 float* __restrict__ hsum,
                                                 float* __restrict__ hasobs) {
  __shared__ _Float16 sA[16 * 256];   // E^T staged (rows 4..15 zero)
  __shared__ _Float16 sH[256 * CD];   // h_out chunk (256 j x 32 d)
  __shared__ float sred[8 * 4];
  __shared__ float smax[4], ssum[4];
  __shared__ float sC[8][16 * CD];

  const int t = threadIdx.x, lane = t & 31, wave = t >> 5;
  const int b = blockIdx.x;
  const int nlo = lane & 15;
  const int hih = lane >> 4;

  for (int i = 4 * 256 + t; i < 16 * 256; i += 256) sA[i] = (_Float16)0.f;

  // phase 1: per-head max over J
  float m0 = -3.4e38f, m1 = m0, m2 = m0, m3 = m0;
  for (int j = t; j < CJ; j += 256) {
    float4 l = *(const float4*)(logitsG + ((size_t)b * CJ + j) * 4);
    m0 = fmaxf(m0, l.x); m1 = fmaxf(m1, l.y);
    m2 = fmaxf(m2, l.z); m3 = fmaxf(m3, l.w);
  }
  m0 = wred_max(m0); m1 = wred_max(m1); m2 = wred_max(m2); m3 = wred_max(m3);
  if (lane == 0) {
    sred[wave * 4 + 0] = m0; sred[wave * 4 + 1] = m1;
    sred[wave * 4 + 2] = m2; sred[wave * 4 + 3] = m3;
  }
  __syncthreads();
  if (t < 4) {
    float m = sred[t];
    for (int wv = 1; wv < 8; ++wv) m = fmaxf(m, sred[wv * 4 + t]);
    smax[t] = m;
  }
  __syncthreads();
  const float mx0 = smax[0], mx1 = smax[1], mx2 = smax[2], mx3 = smax[3];

  // phase 2: exp + WMMA accumulation over J
  float se0 = 0.f, se1 = 0.f, se2 = 0.f, se3 = 0.f;
  v8f c0 = {}; v8f c1 = {};
  const unsigned ldsH = (unsigned)(uintptr_t)(&sH[0]) + (unsigned)t * 64u;

  for (int ch = 0; ch < CJ / 256; ++ch) {
    const int j = ch * 256 + t;
    float4 l = *(const float4*)(logitsG + ((size_t)b * CJ + j) * 4);
    float e0 = (l.x > -1e8f) ? __expf(l.x - mx0) : 0.f;
    float e1 = (l.y > -1e8f) ? __expf(l.y - mx1) : 0.f;
    float e2 = (l.z > -1e8f) ? __expf(l.z - mx2) : 0.f;
    float e3 = (l.w > -1e8f) ? __expf(l.w - mx3) : 0.f;
    se0 += e0; se1 += e1; se2 += e2; se3 += e3;
    sA[0 * 256 + t] = (_Float16)e0;
    sA[1 * 256 + t] = (_Float16)e1;
    sA[2 * 256 + t] = (_Float16)e2;
    sA[3 * 256 + t] = (_Float16)e3;

    // async copy h_out chunk (16 KB contiguous) to LDS; offset applies to
    // both the LDS and the global address per the TDM/async pseudocode.
    const uint64_t gsrc = (uint64_t)(uintptr_t)(
        (const char*)(houtG + ((size_t)b * CJ + (size_t)ch * 256) * CD) +
        (size_t)t * 64);
    asm volatile("global_load_async_to_lds_b128 %0, %1, off"
                 :: "v"(ldsH), "v"(gsrc) : "memory");
    asm volatile("global_load_async_to_lds_b128 %0, %1, off offset:16"
                 :: "v"(ldsH), "v"(gsrc) : "memory");
    asm volatile("global_load_async_to_lds_b128 %0, %1, off offset:32"
                 :: "v"(ldsH), "v"(gsrc) : "memory");
    asm volatile("global_load_async_to_lds_b128 %0, %1, off offset:48"
                 :: "v"(ldsH), "v"(gsrc) : "memory");
    asm volatile("s_wait_asynccnt 0" ::: "memory");
    __syncthreads();

    {  // each wave takes K-chunk kc == wave (8 waves x 32 j = 256 j)
      const int kc = wave;
      const _Float16* ap = sA + nlo * 256 + kc * 32 + hih * 16;
      v8h lo = *(const v8h*)ap;
      v8h hh = *(const v8h*)(ap + 8);
      v16h a;
#pragma unroll
      for (int i = 0; i < 8; ++i) { a[i] = lo[i]; a[8 + i] = hh[i]; }
      v16h b0, b1;
      const int kbb = kc * 32 + hih * 16;
#pragma unroll
      for (int i = 0; i < 16; ++i) {
        b0[i] = sH[(kbb + i) * CD + nlo];
        b1[i] = sH[(kbb + i) * CD + 16 + nlo];
      }
      c0 = wmma16(a, b0, c0);
      c1 = wmma16(a, b1, c1);
    }
    __syncthreads();
  }

  se0 = wred_add(se0); se1 = wred_add(se1);
  se2 = wred_add(se2); se3 = wred_add(se3);
  if (lane == 0) {
    sred[wave * 4 + 0] = se0; sred[wave * 4 + 1] = se1;
    sred[wave * 4 + 2] = se2; sred[wave * 4 + 3] = se3;
  }
  const int mo = hih * 8;
#pragma unroll
  for (int v = 0; v < 8; ++v) {
    sC[wave][(mo + v) * CD + nlo]      = c0[v];
    sC[wave][(mo + v) * CD + 16 + nlo] = c1[v];
  }
  __syncthreads();
  if (t < 4) {
    float s = 0.f;
    for (int wv = 0; wv < 8; ++wv) s += sred[wv * 4 + t];
    ssum[t] = s;
  }
  __syncthreads();
  if (t < 128) {
    const int w = t >> 5, d = t & 31;
    float acc = 0.f;
    for (int wv = 0; wv < 8; ++wv) acc += sC[wv][w * CD + d];
    hsum[(size_t)b * 128 + t] = acc / fmaxf(ssum[w], 1e-30f);
  }
  if (t == 0) hasobs[b] = (ssum[0] > 0.f) ? 1.f : 0.f;
}

// ---------------------------------------------------------------------------
// K4: combiner + 2-layer encoder over 64 rows (tiny), single workgroup
// ---------------------------------------------------------------------------
__global__ __launch_bounds__(256) void k4_final(
    const float* __restrict__ hsum, const float* __restrict__ hasobs,
    const float* __restrict__ c_w, const float* __restrict__ c_b,
    const float* __restrict__ clng, const float* __restrict__ clnb,
    const float* __restrict__ e_w1, const float* __restrict__ e_b1,
    const float* __restrict__ l1g, const float* __restrict__ l1b,
    const float* __restrict__ e_w2, const float* __restrict__ e_b2,
    const float* __restrict__ l2g, const float* __restrict__ l2b,
    float* __restrict__ out) {
  __shared__ float spre[CB * CD];
  __shared__ float scomb[CB * CD];
  __shared__ float sE[CB * CEH];
  __shared__ float sF[CB * 64];
  __shared__ float smu[CB], srs[CB];
  const int t = threadIdx.x;

  for (int o = t; o < CB * CD; o += 256) {
    const int row = o >> 5, d = o & 31;
    float acc = c_b[d];
    for (int k = 0; k < CH; ++k)
      acc = fmaf(hsum[row * CH + k], c_w[k * CD + d], acc);
    spre[o] = acc;
  }
  __syncthreads();
  if (t < CB) {
    float s = 0.f, s2 = 0.f;
    for (int d = 0; d < CD; ++d) { float v = spre[t * CD + d]; s += v; s2 += v * v; }
    float mean = s / CD; smu[t] = mean;
    srs[t] = rsqrtf(s2 / CD - mean * mean + 1e-5f);
  }
  __syncthreads();
  for (int o = t; o < CB * CD; o += 256) {
    const int row = o >> 5, d = o & 31;
    float v = fmaxf((spre[o] - smu[row]) * srs[row] * clng[d] + clnb[d], 0.f);
    scomb[o] = v * hasobs[row];
  }
  __syncthreads();
  for (int o = t; o < CB * CEH; o += 256) {
    const int row = o >> 8, e = o & 255;
    float acc = e_b1[e];
    for (int k = 0; k < CD; ++k)
      acc = fmaf(scomb[row * CD + k], e_w1[k * CEH + e], acc);
    sE[o] = acc;
  }
  __syncthreads();
  if (t < CB) {
    float s = 0.f, s2 = 0.f;
    for (int e = 0; e < CEH; ++e) { float v = sE[t * CEH + e]; s += v; s2 += v * v; }
    float mean = s / CEH; smu[t] = mean;
    srs[t] = rsqrtf(s2 / CEH - mean * mean + 1e-5f);
  }
  __syncthreads();
  for (int o = t; o < CB * CEH; o += 256) {
    const int row = o >> 8, e = o & 255;
    sE[o] = fmaxf((sE[o] - smu[row]) * srs[row] * l1g[e] + l1b[e], 0.f);
  }
  __syncthreads();
  for (int o = t; o < CB * 64; o += 256) {
    const int row = o >> 6, q = o & 63;
    float acc = e_b2[q];
    for (int k = 0; k < CEH; ++k)
      acc = fmaf(sE[row * CEH + k], e_w2[k * 64 + q], acc);
    sF[o] = acc;
  }
  __syncthreads();
  if (t < CB) {
    float s = 0.f, s2 = 0.f;
    for (int q = 0; q < 64; ++q) { float v = sF[t * 64 + q]; s += v; s2 += v * v; }
    float mean = s / 64.f; smu[t] = mean;
    srs[t] = rsqrtf(s2 / 64.f - mean * mean + 1e-5f);
  }
  __syncthreads();
  for (int o = t; o < CB * 64; o += 256) {
    const int row = o >> 6, q = o & 63;
    float v = fmaxf((sF[o] - smu[row]) * srs[row] * l2g[q] + l2b[q], 0.f);
    if (q < CL) out[row * CL + q] = v;               // mu
    else        out[CB * CL + row * CL + (q - CL)] = v;  // logvar
  }
}

// ---------------------------------------------------------------------------
// workspace layout (bytes)
// ---------------------------------------------------------------------------
static constexpr size_t WS_P      = 0;                                    // 8 MiB fp32
static constexpr size_t WS_HOUT   = (size_t)CJ * CH * 4;                  // +64 MiB f16
static constexpr size_t WS_LOGITS = WS_HOUT + (size_t)CB * CJ * CD * 2;   // +16 MiB fp32
static constexpr size_t WS_HSUM   = WS_LOGITS + (size_t)CB * CJ * 4 * 4;
static constexpr size_t WS_OBS    = WS_HSUM + (size_t)CB * CH * 4;

extern "C" void kernel_launch(void* const* d_in, const int* in_sizes, int n_in,
                              void* d_out, int out_size, void* d_ws, size_t ws_size,
                              hipStream_t stream) {
  const float* x        = (const float*)d_in[0];
  const int*   mask     = (const int*)  d_in[1];
  const float* feat_emb = (const float*)d_in[2];
  const float* h_w1     = (const float*)d_in[3];
  const float* h_b1     = (const float*)d_in[4];
  const float* ln1g     = (const float*)d_in[5];
  const float* ln1b     = (const float*)d_in[6];
  const float* h_w2     = (const float*)d_in[7];
  const float* h_b2     = (const float*)d_in[8];
  const float* ln2g     = (const float*)d_in[9];
  const float* ln2b     = (const float*)d_in[10];
  const float* g_w1     = (const float*)d_in[11];
  const float* g_b1     = (const float*)d_in[12];
  const float* g_w2     = (const float*)d_in[13];
  const float* g_b2     = (const float*)d_in[14];
  const float* c_w      = (const float*)d_in[15];
  const float* c_b      = (const float*)d_in[16];
  const float* clng     = (const float*)d_in[17];
  const float* clnb     = (const float*)d_in[18];
  const float* e_w1     = (const float*)d_in[19];
  const float* e_b1     = (const float*)d_in[20];
  const float* l1g      = (const float*)d_in[21];
  const float* l1b      = (const float*)d_in[22];
  const float* e_w2     = (const float*)d_in[23];
  const float* e_b2     = (const float*)d_in[24];
  const float* l2g      = (const float*)d_in[25];
  const float* l2b      = (const float*)d_in[26];

  char* ws = (char*)d_ws;
  float*    P      = (float*)(ws + WS_P);
  _Float16* hout   = (_Float16*)(ws + WS_HOUT);
  float*    logits = (float*)(ws + WS_LOGITS);
  float*    hs     = (float*)(ws + WS_HSUM);
  float*    obs    = (float*)(ws + WS_OBS);

  k1_proj<<<CJ / 128, 256, 0, stream>>>(feat_emb, h_w1, h_b1, P);
  k2_rows<<<dim3(CJ / 128, CB), 256, 0, stream>>>(x, mask, P, h_w1, ln1g, ln1b,
                                                  h_w2, h_b2, ln2g, ln2b, g_w1,
                                                  g_b1, g_w2, g_b2, hout, logits);
  k3_reduce<<<CB, 256, 0, stream>>>(logits, hout, hs, obs);
  k4_final<<<1, 256, 0, stream>>>(hs, obs, c_w, c_b, clng, clnb, e_w1, e_b1,
                                  l1g, l1b, e_w2, e_b2, l2g, l2b, (float*)d_out);
  (void)in_sizes; (void)n_in; (void)out_size; (void)ws_size;
}
